// ConditionalDiffusion_13151189861032
// MI455X (gfx1250) — compile-verified, matrix operand order checked by reading, not run on hardware
//
#include <hip/hip_runtime.h>
#include <hip/hip_bf16.h>

// ---------------------------------------------------------------------------
// Types for CDNA5 WMMA (wave32): D(16x16 f32) = A(16x32 bf16) * B(32x16 bf16) + C
// ---------------------------------------------------------------------------
typedef __attribute__((ext_vector_type(16))) __bf16 v16bf;
typedef __attribute__((ext_vector_type(8)))  float  v8f;

// K-index mapping for 16-bit A/B fragments (CDNA5 ISA 7.12.2):
// lanes 0-15:  element e -> K = (e<8 ? e : e+8)        (K in {0..7,16..23})
// lanes 16-31: element e -> K = 8 + (e<8 ? e : e+8)    (K in {8..15,24..31})
__device__ __host__ __forceinline__ int kmap(int e, int lane) {
  return ((lane >> 4) << 3) + (e < 8 ? e : e + 8);
}

__device__ __forceinline__ v8f wmma_bf16(v16bf a, v16bf b, v8f c) {
  return __builtin_amdgcn_wmma_f32_16x16x32_bf16(false, a, false, b, (short)0, c,
                                                 false, false);
}

// ---------------------------------------------------------------------------
// Tensor Data Mover: linear global->LDS copy of `ndw` dwords.
// D# per CDNA5 ISA 8.3/8.4: group0 = {count, lds_addr, global_addr, type=2},
// group1 = {data_size=4B, tensor_dim0 = tile_dim0 = ndw, stride0 = ndw}.
// ---------------------------------------------------------------------------
#if defined(__has_builtin)
#if __has_builtin(__builtin_amdgcn_tensor_load_to_lds) && \
    __has_builtin(__builtin_amdgcn_s_wait_tensorcnt)
#define HAVE_TDM 1
#endif
#endif
#ifndef HAVE_TDM
#define HAVE_TDM 0
#endif

#if HAVE_TDM
typedef __attribute__((ext_vector_type(4))) unsigned int v4u_t;
typedef __attribute__((ext_vector_type(8))) int          v8i_t;
typedef __attribute__((ext_vector_type(4))) int          v4i_t;

__device__ __forceinline__ void tdm_load_linear(unsigned lds_addr, const void* gsrc,
                                                unsigned ndw) {
  unsigned long long ga = (unsigned long long)(uintptr_t)gsrc;
  v4u_t g0;
  g0[0] = 1u;                                              // count=1 (valid D#)
  g0[1] = lds_addr;                                        // LDS byte address
  g0[2] = (unsigned)(ga & 0xffffffffu);                    // global_addr[31:0]
  g0[3] = (unsigned)((ga >> 32) & 0x01ffffffu) | (2u << 30); // addr[56:32], type=2
  v8i_t g1;
  g1[0] = (int)(2u << 16);                                 // data_size=4B, mask=0
  g1[1] = (int)((ndw & 0xffffu) << 16);                    // tensor_dim0[15:0]
  g1[2] = (int)(((ndw >> 16) & 0xffffu) | (1u << 16));     // dim0[31:16], dim1=1
  g1[3] = (int)((ndw & 0xffffu) << 16);                    // tile_dim0 = ndw
  g1[4] = 0;                                               // tile_dim1/2 unused
  g1[5] = (int)ndw;                                        // dim0_stride[31:0]
  g1[6] = 0;
  g1[7] = 0;
  v4i_t z4 = {0, 0, 0, 0};
#if __clang_major__ >= 23
  v8i_t z8 = {0, 0, 0, 0, 0, 0, 0, 0};
  __builtin_amdgcn_tensor_load_to_lds(g0, g1, z4, z4, z8, 0);
#else
  __builtin_amdgcn_tensor_load_to_lds(g0, g1, z4, z4, 0);
#endif
}
#endif

// ---------------------------------------------------------------------------
// Problem constants
// ---------------------------------------------------------------------------
static constexpr int B  = 64;
static constexpr int HW = 128;
static constexpr int PLANE = HW * HW;   // 16384 = 1<<14
static constexpr int FFN = 2048;
static constexpr int ENC_K = 3 * PLANE; // 49152
static constexpr int KSPLIT = 16;       // encoder GEMM K-split factor

// Workspace layout (bytes)
__host__ __device__ static constexpr size_t aup(size_t x) { return (x + 255) & ~(size_t)255; }
static constexpr size_t OFF_ABAR  = 0;                                         // 1000 f32
static constexpr size_t OFF_PART  = aup(OFF_ABAR + 1000 * 4);                  // 8192 f32
static constexpr size_t OFF_X0    = aup(OFF_PART + 8192 * 4);                  // 64x128 f32
static constexpr size_t OFF_XPART = aup(OFF_X0 + 64 * 128 * 4);                // 16x64x128 f32
static constexpr size_t OFF_MID   = aup(OFF_XPART + (size_t)KSPLIT * 8192 * 4);// 64x64 f32
static constexpr size_t OFF_W1    = aup(OFF_MID + 64 * 64 * 4);                // 32x32  bf16 packed
static constexpr size_t OFF_W2    = aup(OFF_W1 + 32 * 32 * 2);                 // 288x64 bf16 packed
static constexpr size_t OFF_W3    = aup(OFF_W2 + 288 * 64 * 2);                // 576x32 bf16 packed
static constexpr size_t OFF_W4    = aup(OFF_W3 + 576 * 32 * 2);                // 288x16 bf16 packed
static constexpr size_t OFF_ENCT  = aup(OFF_W4 + 288 * 16 * 2);                // 49152x128 bf16 packed
static constexpr size_t OFF_ZT    = aup(OFF_ENCT + (size_t)ENC_K * 128 * 2);
static constexpr size_t OFF_H1    = aup(OFF_ZT + (size_t)B * 3 * PLANE * 2);
static constexpr size_t OFF_H2    = aup(OFF_H1 + (size_t)B * 32 * PLANE * 2);
static constexpr size_t OFF_H3    = aup(OFF_H2 + (size_t)B * 64 * PLANE * 2);

// ---------------------------------------------------------------------------
// Prep: alpha_bars cumprod (LinearScheduler)
// ---------------------------------------------------------------------------
__global__ void prep_abars(float* abars) {
  if (threadIdx.x == 0 && blockIdx.x == 0) {
    float ab = 1.f;
    for (int j = 0; j < 1000; ++j) {
      float beta = 1e-4f + (0.02f - 1e-4f) * (float)j / 999.f;
      ab *= (1.f - beta);
      abars[j] = ab;
    }
  }
}

// enc_lin_w [128][49152] f32 -> bf16 packed B-fragments [kt][ct(8)][lane(32)][e(16)]
__global__ void cvt_enc(const float* __restrict__ src, __bf16* __restrict__ dst) {
  unsigned j = blockIdx.x * 256u + threadIdx.x;   // 24576*256 == 49152*128 exactly
  int e = j & 15, lane = (j >> 4) & 31;
  unsigned f = j >> 9;
  int ct = f & 7, kt = f >> 3;
  int k = (kt << 5) + kmap(e, lane);
  int n = (ct << 4) + (lane & 15);
  dst[j] = (__bf16)src[(size_t)n * ENC_K + k];
}

// conv weight [CO][CI][3][3] f32 -> bf16 packed B-fragments
// layout [kt][ct(NCT)][lane(32)][e(16)], zero padded (k>=CI*9 or co>=CO)
__global__ void cvt_conv_w(const float* __restrict__ src, __bf16* __restrict__ dst,
                           int CO, int CI, int KPAD, int COLS) {
  int j = blockIdx.x * 256 + threadIdx.x;
  if (j >= KPAD * COLS) return;
  int NCT = COLS >> 4;
  int e = j & 15, lane = (j >> 4) & 31, f = j >> 9;
  int ct = f % NCT, kt = f / NCT;
  int k = (kt << 5) + kmap(e, lane);
  int co = (ct << 4) + (lane & 15);
  float v = 0.f;
  if (k < CI * 9 && co < CO) {
    int ci = k / 9;
    v = src[(co * CI + ci) * 9 + (k - ci * 9)];
  }
  dst[j] = (__bf16)v;
}

// q_sample: z_t = sqrt(ab)*tgt + sqrt(1-ab)*noise   (bf16 output)
__global__ void qsample_kernel(const float* __restrict__ tgt, const float* __restrict__ noise,
                               const int* __restrict__ t, const float* __restrict__ abars,
                               __bf16* __restrict__ zt) {
  int i = blockIdx.x * 256 + threadIdx.x;          // 12288*256 == 64*3*16384 exactly
  int n = i / (3 * PLANE);
  float ab = abars[t[n]];
  float z = sqrtf(ab) * tgt[i] + sqrtf(1.f - ab) * noise[i];
  zt[i] = (__bf16)z;
}

// ---------------------------------------------------------------------------
// Encoder input GEMM: xpart[kc,64,128] partial of cond[64,49152] @ W.T
// grid (4,8,KSPLIT) 16x16 tiles, one wave each, 96 WMMA steps per block.
// A loaded as 4x float4 (32B-aligned contiguous K runs), B as packed v16bf.
// ---------------------------------------------------------------------------
__global__ void enc_gemm(const float* __restrict__ cond, const __bf16* __restrict__ wTp,
                         float* __restrict__ xpart) {
  const int lane = threadIdx.x;
  const int bm = blockIdx.x, bn = blockIdx.y, kc = blockIdx.z;
  constexpr int KSTEPS = ENC_K / 32 / KSPLIT;      // 96
  const int khalf = (lane >> 4) << 3;              // 0 or 8
  const int m = (bm << 4) + (lane & 15);
  const float* arow = cond + (size_t)m * ENC_K;
  const v16bf* wfrag = (const v16bf*)wTp;
  v8f acc = {0.f, 0.f, 0.f, 0.f, 0.f, 0.f, 0.f, 0.f};
  for (int kt = kc * KSTEPS; kt < (kc + 1) * KSTEPS; ++kt) {
    const float4* a4 = (const float4*)(arow + (kt << 5) + khalf);
    float4 p0 = a4[0], p1 = a4[1];                 // K = base .. base+7
    float4 q0 = a4[4], q1 = a4[5];                 // K = base+16 .. base+23
    v16bf af;
    af[0] = (__bf16)p0.x; af[1] = (__bf16)p0.y; af[2]  = (__bf16)p0.z; af[3]  = (__bf16)p0.w;
    af[4] = (__bf16)p1.x; af[5] = (__bf16)p1.y; af[6]  = (__bf16)p1.z; af[7]  = (__bf16)p1.w;
    af[8] = (__bf16)q0.x; af[9] = (__bf16)q0.y; af[10] = (__bf16)q0.z; af[11] = (__bf16)q0.w;
    af[12] = (__bf16)q1.x; af[13] = (__bf16)q1.y; af[14] = (__bf16)q1.z; af[15] = (__bf16)q1.w;
    v16bf bfm = wfrag[((kt << 3) + bn) * 32 + lane];
    acc = wmma_bf16(af, bfm, acc);
  }
  const int n = (bn << 4) + (lane & 15);
#pragma unroll
  for (int r = 0; r < 8; ++r) {
    int mm = (bm << 4) + r + khalf;
    xpart[(kc << 13) + (mm << 7) + n] = acc[r];
  }
}

// Deterministic fixed-order K-split reduction + bias
__global__ void enc_reduce(const float* __restrict__ xpart, const float* __restrict__ bias,
                           float* __restrict__ x0) {
  int i = blockIdx.x * 256 + threadIdx.x;          // 32*256 == 8192
  float s = bias[i & 127];
  for (int kc = 0; kc < KSPLIT; ++kc) s += xpart[(kc << 13) + i];
  x0[i] = s;
}

// ---------------------------------------------------------------------------
// Fused transformer encoder (seq_len==1 => attn == out_proj(v_proj(x))),
// 2 layers + LayerNorms + FFN, plus midadd[64,64] = c @ mid_w.T + mid_b.
// Tiny (<0.2% of FLOPs): single workgroup, plain VALU.
// ---------------------------------------------------------------------------
__global__ void __launch_bounds__(256)
encoder_transformer(const float* __restrict__ x0,
                    const float* __restrict__ in_w, const float* __restrict__ in_b,
                    const float* __restrict__ out_w, const float* __restrict__ out_b,
                    const float* __restrict__ ff1_w, const float* __restrict__ ff1_b,
                    const float* __restrict__ ff2_w, const float* __restrict__ ff2_b,
                    const float* __restrict__ ln1_g, const float* __restrict__ ln1_b,
                    const float* __restrict__ ln2_g, const float* __restrict__ ln2_b,
                    const float* __restrict__ mid_w, const float* __restrict__ mid_b,
                    float* __restrict__ midadd) {
  __shared__ float xs[64 * 128];      // 32 KB
  __shared__ float hs[8 * FFN];       // 64 KB (aliased: v = hs[0:8192), a = hs[8192:16384))
  const int NT = 256;
  int tid = threadIdx.x;
  for (int i = tid; i < 64 * 128; i += NT) xs[i] = x0[i];
  __syncthreads();

  for (int li = 0; li < 2; ++li) {
    const float* wv = in_w + li * 384 * 128 + 256 * 128;  // v-proj rows [2D,3D)
    const float* bv = in_b + li * 384 + 256;
    const float* wo = out_w + li * 128 * 128;
    const float* bo = out_b + li * 128;
    float* vbuf = hs;
    float* abuf = hs + 8192;
    for (int i = tid; i < 8192; i += NT) {          // v = x @ wv.T + bv
      int r = i >> 7, c = i & 127;
      float s = bv[c];
      const float* wr = wv + (c << 7);
      const float* xr = xs + (r << 7);
      for (int k = 0; k < 128; ++k) s += xr[k] * wr[k];
      vbuf[i] = s;
    }
    __syncthreads();
    for (int i = tid; i < 8192; i += NT) {          // a = v @ wo.T + bo
      int r = i >> 7, c = i & 127;
      float s = bo[c];
      const float* wr = wo + (c << 7);
      const float* vr = vbuf + (r << 7);
      for (int k = 0; k < 128; ++k) s += vr[k] * wr[k];
      abuf[i] = s;
    }
    __syncthreads();
    if (tid < 64) {                                 // x = LN1(x + a)
      float* xr = xs + (tid << 7);
      const float* ar = abuf + (tid << 7);
      float m = 0.f;
      for (int k = 0; k < 128; ++k) { xr[k] += ar[k]; m += xr[k]; }
      m *= (1.f / 128.f);
      float v2 = 0.f;
      for (int k = 0; k < 128; ++k) { float d = xr[k] - m; v2 += d * d; }
      float inv = rsqrtf(v2 * (1.f / 128.f) + 1e-5f);
      const float* g = ln1_g + li * 128; const float* bb = ln1_b + li * 128;
      for (int k = 0; k < 128; ++k) xr[k] = (xr[k] - m) * inv * g[k] + bb[k];
    }
    __syncthreads();
    const float* w1 = ff1_w + (size_t)li * FFN * 128;
    const float* b1 = ff1_b + li * FFN;
    const float* w2 = ff2_w + (size_t)li * 128 * FFN;
    const float* b2 = ff2_b + li * 128;
    for (int rc = 0; rc < 8; ++rc) {                // FFN, 8-row chunks
      int r0 = rc * 8;
      for (int i = tid; i < 8 * FFN; i += NT) {
        int r = r0 + (i >> 11), h = i & (FFN - 1);
        float s = b1[h];
        const float* wr = w1 + (size_t)h * 128;
        const float* xr = xs + (r << 7);
        for (int k = 0; k < 128; ++k) s += xr[k] * wr[k];
        hs[i] = fmaxf(s, 0.f);
      }
      __syncthreads();
      for (int i = tid; i < 8 * 128; i += NT) {
        int rr = i >> 7, c = i & 127;
        float s = b2[c];
        const float* wr = w2 + (size_t)c * FFN;
        const float* hr = hs + (rr << 11);
        for (int k = 0; k < FFN; ++k) s += hr[k] * wr[k];
        xs[((r0 + rr) << 7) + c] += s;
      }
      __syncthreads();
      if (tid < 8) {                                // LN2 on this chunk
        float* xr = xs + ((r0 + tid) << 7);
        float m = 0.f;
        for (int k = 0; k < 128; ++k) m += xr[k];
        m *= (1.f / 128.f);
        float v2 = 0.f;
        for (int k = 0; k < 128; ++k) { float d = xr[k] - m; v2 += d * d; }
        float inv = rsqrtf(v2 * (1.f / 128.f) + 1e-5f);
        const float* g = ln2_g + li * 128; const float* bb = ln2_b + li * 128;
        for (int k = 0; k < 128; ++k) xr[k] = (xr[k] - m) * inv * g[k] + bb[k];
      }
      __syncthreads();
    }
  }
  for (int i = tid; i < 64 * 64; i += NT) {         // midadd = c @ mid_w.T + mid_b
    int n = i >> 6, j = i & 63;
    float s = mid_b[j];
    const float* wr = mid_w + (j << 7);
    const float* xr = xs + (n << 7);
    for (int k = 0; k < 128; ++k) s += xr[k] * wr[k];
    midadd[i] = s;
  }
}

// ---------------------------------------------------------------------------
// conv3x3 as implicit GEMM on WMMA bf16.
//   M = 16 x-positions, N = Cout, K = Cin*9 (padded to KPAD, multiple of 32).
//   One workgroup per (n, y). Weight panel (packed B-fragments) is DMA'd to
//   LDS by the Tensor Data Mover while the block stages the 3-row input halo;
//   then s_wait_tensorcnt + barrier. Wave w owns Cout tile w.
// ---------------------------------------------------------------------------
template <int CIN, int COUT, int KPAD, bool RELU, bool MID>
__global__ void conv3x3_wmma(const __bf16* __restrict__ in, const __bf16* __restrict__ wt,
                             const float* __restrict__ bias, const float* __restrict__ midadd,
                             __bf16* __restrict__ out) {
  constexpr int NCT = COUT / 16;
  __shared__ __bf16 a_lds[CIN * 3 * 130];
  __shared__ v16bf w_lds[KPAD * COUT / 16];        // packed fragments, 32B aligned
  const int n = blockIdx.x >> 7;
  const int y = blockIdx.x & 127;
  const int tid = threadIdx.x, nthr = blockDim.x;

#if HAVE_TDM
  if ((tid >> 5) == 0)
    tdm_load_linear((unsigned)(uintptr_t)w_lds, wt, KPAD * COUT / 2);
#else
  {
    __bf16* wb = (__bf16*)w_lds;
    for (int i = tid; i < KPAD * COUT; i += nthr) wb[i] = wt[i];
  }
#endif
  for (int i = tid; i < CIN * 3 * 130; i += nthr) {
    int col = i % 130;
    int rr  = (i / 130) % 3;
    int ci  = i / 390;
    int x = col - 1, yy = y + rr - 1;
    __bf16 v = (__bf16)0.f;
    if (x >= 0 && x < 128 && yy >= 0 && yy < 128)
      v = in[((size_t)(n * CIN + ci) << 14) + (yy << 7) + x];
    a_lds[i] = v;
  }
#if HAVE_TDM
  if ((tid >> 5) == 0) __builtin_amdgcn_s_wait_tensorcnt(0);
#endif
  __syncthreads();

  const int lane = tid & 31;
  const int wv = tid >> 5;
  const int co = (wv << 4) + (lane & 15);
  const float badd = bias[co];
  const float ma = MID ? midadd[(n << 6) + co] : 0.f;

  for (int xt = 0; xt < 8; ++xt) {
    v8f acc = {0.f, 0.f, 0.f, 0.f, 0.f, 0.f, 0.f, 0.f};
    const int xx = (xt << 4) + (lane & 15);        // A row == x position in tile
#pragma unroll
    for (int kt = 0; kt < KPAD / 32; ++kt) {
      v16bf af;
#pragma unroll
      for (int e = 0; e < 16; ++e) {
        int k = (kt << 5) + kmap(e, lane);
        int ci = k / 9, rem = k - ci * 9;
        int r = rem / 3, cx = rem - r * 3;
        af[e] = (k < CIN * 9) ? a_lds[(ci * 3 + r) * 130 + xx + cx] : (__bf16)0.f;
      }
      v16bf bfm = w_lds[(kt * NCT + wv) * 32 + lane];
      acc = wmma_bf16(af, bfm, acc);
    }
#pragma unroll
    for (int r = 0; r < 8; ++r) {
      int xo = (xt << 4) + r + ((lane >> 4) << 3);
      float v = acc[r] + badd;
      if (RELU) v = fmaxf(v, 0.f);
      v += ma;
      out[((size_t)(n * COUT + co) << 14) + (y << 7) + xo] = (__bf16)v;
    }
  }
}

// ---------------------------------------------------------------------------
// Final conv (32 -> 3) fused with MSE partials. COUT padded to 16 (one wave).
// Deterministic: per-block partial sums, fixed-order reduce afterwards.
// ---------------------------------------------------------------------------
__global__ void conv4_loss(const __bf16* __restrict__ in, const __bf16* __restrict__ wt,
                           const float* __restrict__ bias, const float* __restrict__ noise,
                           float* __restrict__ partials) {
  constexpr int CIN = 32, KPAD = 288, COUT = 16;
  __shared__ __bf16 a_lds[CIN * 3 * 130];
  __shared__ v16bf w_lds[KPAD * COUT / 16];
  const int n = blockIdx.x >> 7;
  const int y = blockIdx.x & 127;
  const int lane = threadIdx.x;

#if HAVE_TDM
  tdm_load_linear((unsigned)(uintptr_t)w_lds, wt, KPAD * COUT / 2);
#else
  {
    __bf16* wb = (__bf16*)w_lds;
    for (int i = lane; i < KPAD * COUT; i += 32) wb[i] = wt[i];
  }
#endif
  for (int i = lane; i < CIN * 3 * 130; i += 32) {
    int col = i % 130;
    int rr  = (i / 130) % 3;
    int ci  = i / 390;
    int x = col - 1, yy = y + rr - 1;
    __bf16 v = (__bf16)0.f;
    if (x >= 0 && x < 128 && yy >= 0 && yy < 128)
      v = in[((size_t)(n * CIN + ci) << 14) + (yy << 7) + x];
    a_lds[i] = v;
  }
#if HAVE_TDM
  __builtin_amdgcn_s_wait_tensorcnt(0);
#endif
  __syncthreads();

  const int co = lane & 15;
  float ssum = 0.f;
  for (int xt = 0; xt < 8; ++xt) {
    v8f acc = {0.f, 0.f, 0.f, 0.f, 0.f, 0.f, 0.f, 0.f};
    const int xx = (xt << 4) + (lane & 15);
#pragma unroll
    for (int kt = 0; kt < KPAD / 32; ++kt) {
      v16bf af;
#pragma unroll
      for (int e = 0; e < 16; ++e) {
        int k = (kt << 5) + kmap(e, lane);
        int ci = k / 9, rem = k - ci * 9;
        int r = rem / 3, cx = rem - r * 3;
        af[e] = a_lds[(ci * 3 + r) * 130 + xx + cx];
      }
      v16bf bfm = w_lds[kt * 32 + lane];
      acc = wmma_bf16(af, bfm, acc);
    }
    if (co < 3) {
      float badd = bias[co];
#pragma unroll
      for (int r = 0; r < 8; ++r) {
        int xo = (xt << 4) + r + ((lane >> 4) << 3);
        float e = acc[r] + badd - noise[((size_t)(n * 3 + co) << 14) + (y << 7) + xo];
        ssum += e * e;
      }
    }
  }
  for (int off = 16; off > 0; off >>= 1) ssum += __shfl_down(ssum, off, 32);
  if (lane == 0) partials[blockIdx.x] = ssum;
}

__global__ void loss_reduce(const float* __restrict__ partials, float* __restrict__ out) {
  __shared__ float sh[256];
  int tid = threadIdx.x;
  float s = 0.f;
  for (int i = tid; i < 8192; i += 256) s += partials[i];
  sh[tid] = s;
  __syncthreads();
  for (int st = 128; st > 0; st >>= 1) {
    if (tid < st) sh[tid] += sh[tid + st];
    __syncthreads();
  }
  if (tid == 0) out[0] = sh[0] * (1.f / (float)(B * 3 * PLANE));
}

// ---------------------------------------------------------------------------
// Host orchestration
// ---------------------------------------------------------------------------
extern "C" void kernel_launch(void* const* d_in, const int* in_sizes, int n_in,
                              void* d_out, int out_size, void* d_ws, size_t ws_size,
                              hipStream_t stream) {
  const float* cond  = (const float*)d_in[0];
  const float* tgt   = (const float*)d_in[1];
  const float* noise = (const float*)d_in[2];
  const int*   tstep = (const int*)d_in[3];
  const float* enc_w = (const float*)d_in[4];
  const float* enc_b = (const float*)d_in[5];
  const float* in_w  = (const float*)d_in[6];
  const float* in_b  = (const float*)d_in[7];
  const float* out_w = (const float*)d_in[8];
  const float* out_b = (const float*)d_in[9];
  const float* ff1_w = (const float*)d_in[10];
  const float* ff1_b = (const float*)d_in[11];
  const float* ff2_w = (const float*)d_in[12];
  const float* ff2_b = (const float*)d_in[13];
  const float* ln1_g = (const float*)d_in[14];
  const float* ln1_b = (const float*)d_in[15];
  const float* ln2_g = (const float*)d_in[16];
  const float* ln2_b = (const float*)d_in[17];
  const float* d1_w  = (const float*)d_in[18];
  const float* d1_b  = (const float*)d_in[19];
  const float* d2_w  = (const float*)d_in[20];
  const float* d2_b  = (const float*)d_in[21];
  const float* mid_w = (const float*)d_in[22];
  const float* mid_b = (const float*)d_in[23];
  const float* u1_w  = (const float*)d_in[24];
  const float* u1_b  = (const float*)d_in[25];
  const float* u2_w  = (const float*)d_in[26];
  const float* u2_b  = (const float*)d_in[27];

  char* ws = (char*)d_ws;
  float*  abars    = (float*)(ws + OFF_ABAR);
  float*  partials = (float*)(ws + OFF_PART);
  float*  x0       = (float*)(ws + OFF_X0);
  float*  xpart    = (float*)(ws + OFF_XPART);
  float*  midadd   = (float*)(ws + OFF_MID);
  __bf16* w1       = (__bf16*)(ws + OFF_W1);
  __bf16* w2       = (__bf16*)(ws + OFF_W2);
  __bf16* w3       = (__bf16*)(ws + OFF_W3);
  __bf16* w4       = (__bf16*)(ws + OFF_W4);
  __bf16* encT     = (__bf16*)(ws + OFF_ENCT);
  __bf16* zt       = (__bf16*)(ws + OFF_ZT);
  __bf16* h1       = (__bf16*)(ws + OFF_H1);
  __bf16* h2       = (__bf16*)(ws + OFF_H2);
  __bf16* h3       = (__bf16*)(ws + OFF_H3);

  // Prep (weights conversion into packed fragment order, schedule, q_sample)
  prep_abars<<<1, 32, 0, stream>>>(abars);
  cvt_enc<<<24576, 256, 0, stream>>>(enc_w, encT);
  cvt_conv_w<<<(32 * 32 + 255) / 256, 256, 0, stream>>>(d1_w, w1, 32, 3, 32, 32);
  cvt_conv_w<<<(288 * 64 + 255) / 256, 256, 0, stream>>>(d2_w, w2, 64, 32, 288, 64);
  cvt_conv_w<<<(576 * 32 + 255) / 256, 256, 0, stream>>>(u1_w, w3, 32, 64, 576, 32);
  cvt_conv_w<<<(288 * 16 + 255) / 256, 256, 0, stream>>>(u2_w, w4, 3, 32, 288, 16);
  qsample_kernel<<<12288, 256, 0, stream>>>(tgt, noise, tstep, abars, zt);

  // Condition encoder (K-split WMMA GEMM + deterministic reduce + tiny fused TF)
  enc_gemm<<<dim3(4, 8, KSPLIT), 32, 0, stream>>>(cond, encT, xpart);
  enc_reduce<<<32, 256, 0, stream>>>(xpart, enc_b, x0);
  encoder_transformer<<<1, 256, 0, stream>>>(x0, in_w, in_b, out_w, out_b,
                                             ff1_w, ff1_b, ff2_w, ff2_b,
                                             ln1_g, ln1_b, ln2_g, ln2_b,
                                             mid_w, mid_b, midadd);

  // UNet (implicit-GEMM WMMA convs, TDM weight staging)
  conv3x3_wmma<3, 32, 32, true, false><<<8192, 64, 0, stream>>>(zt, w1, d1_b, nullptr, h1);
  conv3x3_wmma<32, 64, 288, true, true><<<8192, 128, 0, stream>>>(h1, w2, d2_b, midadd, h2);
  conv3x3_wmma<64, 32, 576, true, false><<<8192, 64, 0, stream>>>(h2, w3, u1_b, nullptr, h3);
  conv4_loss<<<8192, 32, 0, stream>>>(h3, w4, u2_b, noise, partials);
  loss_reduce<<<1, 256, 0, stream>>>(partials, (float*)d_out);
  (void)in_sizes; (void)n_in; (void)out_size; (void)ws_size;
}
